// DEC_62405874811862
// MI455X (gfx1250) — compile-verified, matrix operand order checked by reading, not run on hardware
//
#include <hip/hip_runtime.h>

typedef __attribute__((ext_vector_type(16))) __bf16 bf16x16;
typedef __attribute__((ext_vector_type(8)))  float  floatx8;
typedef __attribute__((ext_vector_type(8)))  unsigned short u16x8;

#define BB 16
#define PP 2048
#define KNN 5
#define EE (BB*PP*KNN)      // 163840 edges
#define NP (BB*PP)          // 32768 points

__device__ __forceinline__ unsigned short f2b(float f) {
  unsigned int u = __float_as_uint(f);
  unsigned int r = u + 0x7fffu + ((u >> 16) & 1u);
  return (unsigned short)(r >> 16);
}
__device__ __forceinline__ __bf16 b16(unsigned short u) {
  union { unsigned short s; __bf16 h; } x; x.s = u; return x.h;
}
__device__ __forceinline__ void top5_insert(float d, int q, float best[5], int bidx[5]) {
  if (d < best[4]) {
    best[4] = d; bidx[4] = q;
#pragma unroll
    for (int s = 4; s > 0; --s) {
      if (best[s] < best[s-1]) {
        float td = best[s]; best[s] = best[s-1]; best[s-1] = td;
        int tq = bidx[s]; bidx[s] = bidx[s-1]; bidx[s-1] = tq;
      }
    }
  }
}

// ---------------- utility kernels ----------------
__global__ void DEC_fill(float* p, int n, float v) {
  int i = blockIdx.x * 256 + threadIdx.x;
  if (i < n) p[i] = v;
}
__global__ void DEC_f32_to_bf16(const float* __restrict__ src, unsigned short* __restrict__ dst, int n) {
  int i = blockIdx.x * 256 + threadIdx.x;
  if (i < n) dst[i] = f2b(src[i]);
}
// convert [K x N] f32 weights to bf16 padded to [Kpad x N] with zero rows
__global__ void DEC_w_pad(const float* __restrict__ src, unsigned short* __restrict__ dst,
                          int K, int Kpad, int N) {
  int i = blockIdx.x * 256 + threadIdx.x;
  if (i >= Kpad * N) return;
  int k = i / N;
  dst[i] = (k < K) ? f2b(src[i]) : (unsigned short)0;
}
// convert f32 [rows x C] to bf16 [Rpad x C], zero-padding extra rows
__global__ void DEC_pad_bf16(const float* __restrict__ src, unsigned short* __restrict__ dst,
                             int rows, int C, int total) {
  int i = blockIdx.x * 256 + threadIdx.x;
  if (i >= total) return;
  dst[i] = (i / C < rows) ? f2b(src[i]) : (unsigned short)0;
}

// ---------------- kNN on raw positions (D=3) ----------------
__global__ void DEC_knn1(const float* __restrict__ pos, int* __restrict__ idxOut) {
  __shared__ float sp[PP * 3];
  int chunks = PP / 256;
  int b = blockIdx.x / chunks;
  int p = (blockIdx.x % chunks) * 256 + threadIdx.x;
  const float* pb = pos + (size_t)b * PP * 3;
  for (int i = threadIdx.x; i < PP * 3; i += 256) sp[i] = pb[i];
  __syncthreads();
  float x = sp[p*3+0], y = sp[p*3+1], z = sp[p*3+2];
  float best[5]; int bidx[5];
#pragma unroll
  for (int s = 0; s < 5; ++s) { best[s] = 3.4e38f; bidx[s] = 0; }
  for (int q = 0; q < PP; ++q) {
    float dx = sp[q*3+0] - x, dy = sp[q*3+1] - y, dz = sp[q*3+2] - z;
    top5_insert(dx*dx + dy*dy + dz*dz, q, best, bidx);
  }
#pragma unroll
  for (int s = 0; s < 5; ++s) idxOut[((size_t)(b*PP + p))*KNN + s] = bidx[s];
}

// ---------------- edge feature gathers ----------------
// e1: [EE x 32] bf16, cols 0..5 = [xi, xj-xi], cols 6..31 = 0 (K padded to 32)
__global__ void DEC_gather1(const float* __restrict__ pos, const int* __restrict__ idx,
                            unsigned short* __restrict__ e1) {
  int e = blockIdx.x * 256 + threadIdx.x;
  if (e >= EE) return;
  int b = e / (PP * KNN);
  int p = (e / KNN) % PP;
  int j = idx[e];
  const float* pi = pos + ((size_t)(b*PP + p)) * 3;
  const float* pj = pos + ((size_t)(b*PP + j)) * 3;
  unsigned short row[32];
#pragma unroll
  for (int c = 0; c < 32; ++c) row[c] = 0;
#pragma unroll
  for (int c = 0; c < 3; ++c) { float xi = pi[c]; row[c] = f2b(xi); row[3+c] = f2b(pj[c] - xi); }
  u16x8* o = (u16x8*)(e1 + (size_t)e * 32);
#pragma unroll
  for (int v = 0; v < 4; ++v) {
    u16x8 t;
#pragma unroll
    for (int i = 0; i < 8; ++i) t[i] = row[v*8 + i];
    o[v] = t;
  }
}
__global__ void DEC_gather2(const float* __restrict__ x1f, const int* __restrict__ idx,
                            unsigned short* __restrict__ e2) {
  long f = (long)blockIdx.x * 256 + threadIdx.x;
  if (f >= (long)EE * 64) return;
  int c = (int)(f & 63);
  long e = f >> 6;
  int b = (int)(e / (PP * KNN));
  int p = (int)((e / KNN) % PP);
  int j = idx[e];
  float xi = x1f[((size_t)(b*PP + p))*64 + c];
  float xj = x1f[((size_t)(b*PP + j))*64 + c];
  e2[(size_t)e*128 + c]      = f2b(xi);
  e2[(size_t)e*128 + 64 + c] = f2b(xj - xi);
}

// ---------------- bf16 WMMA GEMM: C = relu(A@W + bias) ----------------
// K: compile-time, multiple of 32. lda % 8 == 0, N % 64 == 0.
// grid = (N/64, M/32), block = 256 (8 waves, each one 16x16 C tile)
template<int K>
__global__ void DEC_gemm(const unsigned short* __restrict__ A, int lda,
                         const unsigned short* __restrict__ W,
                         const float* __restrict__ bias,
                         float* __restrict__ C, int N, int relu) {
  __shared__ unsigned short sBt[64][32];   // [n local][k local] transposed W tile
  int tid = threadIdx.x;
  int wave = tid >> 5, lane = tid & 31;
  int wm = wave >> 2, wn = wave & 3;
  int l15 = lane & 15, hi = lane >> 4;
  size_t m0 = (size_t)blockIdx.y * 32 + (size_t)wm * 16;
  int nBase = blockIdx.x * 64;
  const unsigned short* arow = A + (m0 + l15) * (size_t)lda;
  int kk = tid >> 3;
  int nn = (tid & 7) * 8;
  floatx8 acc = {};
  for (int k0 = 0; k0 < K; k0 += 32) {
    const u16x8* ap = (const u16x8*)(arow + k0 + hi * 8);
    u16x8 a0 = ap[0];
    u16x8 a1 = ap[2];                      // +16 elements
    __syncthreads();
    u16x8 wv = *(const u16x8*)(W + (size_t)(k0 + kk) * N + nBase + nn);
#pragma unroll
    for (int i = 0; i < 8; ++i) sBt[nn + i][kk] = wv[i];
    __syncthreads();
    if (k0 + 32 < K) {
      __builtin_prefetch(arow + k0 + 32 + hi * 8, 0, 3);
      __builtin_prefetch(W + (size_t)(k0 + 32 + kk) * N + nBase + nn, 0, 3);
    }
    const u16x8* bp = (const u16x8*)(&sBt[wn * 16 + l15][hi * 8]);
    u16x8 b0 = bp[0];
    u16x8 b1 = bp[2];
    bf16x16 a, bfrag;
#pragma unroll
    for (int i = 0; i < 8; ++i) {
      a[i] = b16(a0[i]); a[i+8] = b16(a1[i]);
      bfrag[i] = b16(b0[i]); bfrag[i+8] = b16(b1[i]);
    }
    acc = __builtin_amdgcn_wmma_f32_16x16x32_bf16(false, a, false, bfrag, (short)0, acc, false, false);
  }
  int n = nBase + wn * 16 + l15;
  float bv = bias[n];
#pragma unroll
  for (int r = 0; r < 8; ++r) {
    size_t m = m0 + r + 8 * hi;
    float v = acc[r] + bv;
    if (relu) v = fmaxf(v, 0.0f);
    C[m * (size_t)N + n] = v;
  }
}

// ---------------- BN statistics over rows of H [M x C] ----------------
__global__ void DEC_stats(const float* __restrict__ H, long M, int C,
                          float* __restrict__ sum, float* __restrict__ sumsq) {
  long flat = (long)blockIdx.x * 256 + threadIdx.x;
  long stripes = ((long)gridDim.x * 256) / C;
  int c = (int)(flat % C);
  long r0 = flat / C;
  float s = 0.f, q = 0.f;
  for (long r = r0; r < M; r += stripes) {
    float v = H[r * C + c]; s += v; q += v * v;
  }
  atomicAdd(&sum[c], s);
  atomicAdd(&sumsq[c], q);
}
__global__ void DEC_finalize(const float* sum, const float* sumsq, const float* g, const float* be,
                             float cnt, int C, float* s, float* t) {
  int c = blockIdx.x * 256 + threadIdx.x;
  if (c >= C) return;
  float mu = sum[c] / cnt;
  float var = sumsq[c] / cnt - mu * mu;
  float sc = g[c] * rsqrtf(var + 1e-5f);
  s[c] = sc; t[c] = be[c] - mu * sc;
}
__global__ void DEC_apply(const float* __restrict__ H, const float* s, const float* t,
                          int C, long n, unsigned short* __restrict__ Y) {
  long i = (long)blockIdx.x * 256 + threadIdx.x;
  if (i >= n) return;
  int c = (int)(i % C);
  Y[i] = f2b(H[i] * s[c] + t[c]);
}

// max over k=5 neighbors of BN-affine(h), writes bf16 into cat (and optional f32)
__global__ void DEC_aggregate(const float* __restrict__ H, const float* s, const float* t, int C,
                              unsigned short* __restrict__ catOut, int catStride, int catOff,
                              float* __restrict__ f32Out) {
  long f = (long)blockIdx.x * 256 + threadIdx.x;
  if (f >= (long)NP * C) return;
  int c = (int)(f % C);
  long pt = f / C;
  float sc = s[c], tt = t[c];
  float m = -3.4e38f;
#pragma unroll
  for (int k = 0; k < KNN; ++k) {
    float v = H[(pt * KNN + k) * (size_t)C + c] * sc + tt;
    m = fmaxf(m, v);
  }
  catOut[pt * (size_t)catStride + catOff + c] = f2b(m);
  if (f32Out) f32Out[pt * (size_t)C + c] = m;
}

__global__ void DEC_norms(const float* __restrict__ X, float* __restrict__ out) {
  int i = blockIdx.x * 256 + threadIdx.x;
  if (i >= NP) return;
  const float* x = X + (size_t)i * 64;
  float s = 0.f;
#pragma unroll
  for (int c = 0; c < 64; ++c) s += x[c] * x[c];
  out[i] = s;
}

// ---------------- fused WMMA kNN on 64-dim features ----------------
// X: bf16 [B*P x ldx] (features in cols 0..63), one wave/block, 16 query rows.
__global__ void DEC_knn2(const unsigned short* __restrict__ X, int ldx,
                         const float* __restrict__ norms, int* __restrict__ idxOut) {
  __shared__ float tile[16][17];
  int tilesPerBatch = PP / 16;
  int b  = blockIdx.x / tilesPerBatch;
  int m0 = (blockIdx.x % tilesPerBatch) * 16;
  int lane = threadIdx.x & 31;
  int l15 = lane & 15, hi = lane >> 4;
  const unsigned short* Xb = X + (size_t)b * PP * ldx;
  const float* nb = norms + (size_t)b * PP;

  float rn[8];
#pragma unroll
  for (int r = 0; r < 8; ++r) rn[r] = nb[m0 + r + 8 * hi];

  float best[5]; int bidx[5];
#pragma unroll
  for (int s = 0; s < 5; ++s) { best[s] = 3.4e38f; bidx[s] = 0; }

  for (int n0 = 0; n0 < PP; n0 += 16) {
    floatx8 acc = {};
#pragma unroll
    for (int ks = 0; ks < 2; ++ks) {
      int k0 = ks * 32;
      bf16x16 a, bfrag;
      const u16x8* ap = (const u16x8*)(Xb + (size_t)(m0 + l15) * ldx + k0 + hi * 8);
      const u16x8* bp = (const u16x8*)(Xb + (size_t)(n0 + l15) * ldx + k0 + hi * 8);
      u16x8 a0 = ap[0], a1 = ap[2], b0 = bp[0], b1 = bp[2];
#pragma unroll
      for (int i = 0; i < 8; ++i) {
        a[i] = b16(a0[i]); a[i+8] = b16(a1[i]);
        bfrag[i] = b16(b0[i]); bfrag[i+8] = b16(b1[i]);
      }
      acc = __builtin_amdgcn_wmma_f32_16x16x32_bf16(false, a, false, bfrag, (short)0, acc, false, false);
    }
    float cn = nb[n0 + l15];
#pragma unroll
    for (int r = 0; r < 8; ++r)
      tile[r + 8 * hi][l15] = rn[r] + cn - 2.0f * acc[r];
    __syncthreads();
    if (lane < 16) {
#pragma unroll
      for (int j = 0; j < 16; ++j) top5_insert(tile[lane][j], n0 + j, best, bidx);
    }
    __syncthreads();
  }
  if (lane < 16) {
    int p = m0 + lane;
#pragma unroll
    for (int s = 0; s < 5; ++s) idxOut[((size_t)(b*PP + p))*KNN + s] = bidx[s];
  }
}

// ---------------- global pooling path (l1) ----------------
__global__ void DEC_bminmax(const float* __restrict__ H, float* __restrict__ mx, float* __restrict__ mn) {
  int flat = blockIdx.x * 256 + threadIdx.x;   // B*1024
  if (flat >= BB * 1024) return;
  int b = flat / 1024, c = flat % 1024;
  float vmx = -3.4e38f, vmn = 3.4e38f;
  for (int p = 0; p < PP; ++p) {
    float v = H[((size_t)(b*PP + p)) * 1024 + c];
    vmx = fmaxf(vmx, v); vmn = fminf(vmn, v);
  }
  mx[flat] = vmx; mn[flat] = vmn;
}
__global__ void DEC_pool(const float* sum, const float* sumsq, const float* mx, const float* mn,
                         const float* g, const float* be, float* __restrict__ pooled) {
  int flat = blockIdx.x * 256 + threadIdx.x;   // B*1024
  if (flat >= BB * 1024) return;
  int c = flat % 1024;
  float cnt = (float)NP;
  float mu = sum[c] / cnt;
  float var = sumsq[c] / cnt - mu * mu;
  float sc = g[c] * rsqrtf(var + 1e-5f);
  float tt = be[c] - mu * sc;
  pooled[flat] = (sc >= 0.f) ? sc * mx[flat] + tt : sc * mn[flat] + tt;
}

// ---------------- tiny head helpers ----------------
__global__ void DEC_head_lin(const float* __restrict__ X, const float* __restrict__ W,
                             const float* __restrict__ bias, float* __restrict__ Y,
                             int B, int Kd, int N, int relu) {
  int flat = blockIdx.x * 256 + threadIdx.x;
  if (flat >= B * N) return;
  int b = flat / N, n = flat % N;
  float acc = bias[n];
  for (int k = 0; k < Kd; ++k) acc += X[(size_t)b * Kd + k] * W[(size_t)k * N + n];
  if (relu) acc = fmaxf(acc, 0.f);
  Y[flat] = acc;
}
__global__ void DEC_head_bn(const float* __restrict__ H, const float* g, const float* be,
                            float* __restrict__ Y, int B, int C) {
  int c = blockIdx.x * 256 + threadIdx.x;
  if (c >= C) return;
  float s = 0.f, q = 0.f;
  for (int b = 0; b < B; ++b) { float v = H[(size_t)b * C + c]; s += v; q += v * v; }
  float mu = s / B;
  float var = q / B - mu * mu;
  float sc = g[c] * rsqrtf(var + 1e-5f);
  float tt = be[c] - mu * sc;
  for (int b = 0; b < B; ++b) Y[(size_t)b * C + c] = H[(size_t)b * C + c] * sc + tt;
}

// =====================================================================
extern "C" void kernel_launch(void* const* d_in, const int* in_sizes, int n_in,
                              void* d_out, int out_size, void* d_ws, size_t ws_size,
                              hipStream_t stream) {
  (void)in_sizes; (void)n_in; (void)out_size; (void)ws_size;
  const float* pos  = (const float*)d_in[0];
  const float* c1w0 = (const float*)d_in[1],  *c1b0 = (const float*)d_in[2],
             * c1g0 = (const float*)d_in[3],  *c1e0 = (const float*)d_in[4];
  const float* c1w1 = (const float*)d_in[5],  *c1b1 = (const float*)d_in[6],
             * c1g1 = (const float*)d_in[7],  *c1e1 = (const float*)d_in[8];
  const float* c1w2 = (const float*)d_in[9],  *c1b2 = (const float*)d_in[10],
             * c1g2 = (const float*)d_in[11], *c1e2 = (const float*)d_in[12];
  const float* c2w  = (const float*)d_in[13], *c2b  = (const float*)d_in[14],
             * c2g  = (const float*)d_in[15], *c2e  = (const float*)d_in[16];
  const float* l1w  = (const float*)d_in[17], *l1b  = (const float*)d_in[18],
             * l1g  = (const float*)d_in[19], *l1e  = (const float*)d_in[20];
  const float* m1w  = (const float*)d_in[21], *m1b  = (const float*)d_in[22],
             * m1g  = (const float*)d_in[23], *m1e  = (const float*)d_in[24];
  const float* m2w  = (const float*)d_in[25], *m2b  = (const float*)d_in[26],
             * m2g  = (const float*)d_in[27], *m2e  = (const float*)d_in[28];
  const float* m3w  = (const float*)d_in[29], *m3b  = (const float*)d_in[30];
  float* out = (float*)d_out;

  // workspace carve-up
  char* base = (char*)d_ws;
  size_t off = 0;
  auto alloc = [&](size_t bytes) -> void* {
    void* p = base + off;
    off = (off + bytes + 255) & ~(size_t)255;
    return p;
  };
  unsigned short* e1   = (unsigned short*)alloc((size_t)EE * 32 * 2);
  unsigned short* e2   = (unsigned short*)alloc((size_t)EE * 128 * 2);
  unsigned short* cat  = (unsigned short*)alloc((size_t)NP * 192 * 2);
  float* hbuf          = (float*)alloc((size_t)NP * 1024 * 4);   // also holds EE x {64,128}
  unsigned short* ybuf = (unsigned short*)alloc((size_t)EE * 64 * 2);
  float* x1f           = (float*)alloc((size_t)NP * 64 * 4);
  int*   idx1          = (int*)alloc((size_t)EE * 4);
  int*   idx2          = (int*)alloc((size_t)EE * 4);
  float* nrm           = (float*)alloc((size_t)NP * 4);
  float* sumb          = (float*)alloc(1024 * 4);
  float* sqb           = (float*)alloc(1024 * 4);
  float* sbuf          = (float*)alloc(1024 * 4);
  float* tbuf          = (float*)alloc(1024 * 4);
  float* bmx           = (float*)alloc((size_t)BB * 1024 * 4);
  float* bmn           = (float*)alloc((size_t)BB * 1024 * 4);
  float* pooled        = (float*)alloc((size_t)BB * 1024 * 4);
  unsigned short* poolb= (unsigned short*)alloc((size_t)32 * 1024 * 2);
  float* h1            = (float*)alloc((size_t)32 * 512 * 4);
  float* y1            = (float*)alloc((size_t)BB * 512 * 4);
  unsigned short* y1b  = (unsigned short*)alloc((size_t)32 * 512 * 2);
  float* h2            = (float*)alloc((size_t)32 * 256 * 4);
  float* y2            = (float*)alloc((size_t)BB * 256 * 4);
  unsigned short* wc10 = (unsigned short*)alloc(32  * 64  * 2);   // padded from 6x64
  unsigned short* wc11 = (unsigned short*)alloc(64  * 64  * 2);
  unsigned short* wc12 = (unsigned short*)alloc(64  * 64  * 2);
  unsigned short* wc2b = (unsigned short*)alloc(128 * 128 * 2);
  unsigned short* wl1b = (unsigned short*)alloc(192 * 1024 * 2);
  unsigned short* wm1b = (unsigned short*)alloc((size_t)1024 * 512 * 2);
  unsigned short* wm2b = (unsigned short*)alloc((size_t)512  * 256 * 2);

  // weight conversions to bf16 (c1w0 zero-padded to K=32)
  DEC_w_pad<<<(32*64+255)/256, 256, 0, stream>>>(c1w0, wc10, 6, 32, 64);
  DEC_f32_to_bf16<<<(64*64+255)/256,    256, 0, stream>>>(c1w1, wc11, 64*64);
  DEC_f32_to_bf16<<<(64*64+255)/256,    256, 0, stream>>>(c1w2, wc12, 64*64);
  DEC_f32_to_bf16<<<(128*128+255)/256,  256, 0, stream>>>(c2w,  wc2b, 128*128);
  DEC_f32_to_bf16<<<(192*1024+255)/256, 256, 0, stream>>>(l1w,  wl1b, 192*1024);
  DEC_f32_to_bf16<<<(1024*512+255)/256, 256, 0, stream>>>(m1w,  wm1b, 1024*512);
  DEC_f32_to_bf16<<<(512*256+255)/256,  256, 0, stream>>>(m2w,  wm2b, 512*256);

  // ---- EdgeConv 1 ----
  DEC_knn1<<<BB * (PP/256), 256, 0, stream>>>(pos, idx1);
  DEC_gather1<<<EE/256, 256, 0, stream>>>(pos, idx1, e1);

  // layer 0: 6 -> 64 (K padded to 32)
  DEC_gemm<32><<<dim3(1, EE/32), 256, 0, stream>>>(e1, 32, wc10, c1b0, hbuf, 64, 1);
  DEC_fill<<<4, 256, 0, stream>>>(sumb, 1024, 0.f);
  DEC_fill<<<4, 256, 0, stream>>>(sqb, 1024, 0.f);
  DEC_stats<<<64, 256, 0, stream>>>(hbuf, EE, 64, sumb, sqb);
  DEC_finalize<<<1, 256, 0, stream>>>(sumb, sqb, c1g0, c1e0, (float)EE, 64, sbuf, tbuf);
  DEC_apply<<<(EE*64)/256, 256, 0, stream>>>(hbuf, sbuf, tbuf, 64, (long)EE*64, ybuf);

  // layer 1: 64 -> 64
  DEC_gemm<64><<<dim3(1, EE/32), 256, 0, stream>>>(ybuf, 64, wc11, c1b1, hbuf, 64, 1);
  DEC_fill<<<4, 256, 0, stream>>>(sumb, 1024, 0.f);
  DEC_fill<<<4, 256, 0, stream>>>(sqb, 1024, 0.f);
  DEC_stats<<<64, 256, 0, stream>>>(hbuf, EE, 64, sumb, sqb);
  DEC_finalize<<<1, 256, 0, stream>>>(sumb, sqb, c1g1, c1e1, (float)EE, 64, sbuf, tbuf);
  DEC_apply<<<(EE*64)/256, 256, 0, stream>>>(hbuf, sbuf, tbuf, 64, (long)EE*64, ybuf);

  // layer 2: 64 -> 64, then max over k -> x1 (cols 0..63 of cat, + f32 copy)
  DEC_gemm<64><<<dim3(1, EE/32), 256, 0, stream>>>(ybuf, 64, wc12, c1b2, hbuf, 64, 1);
  DEC_fill<<<4, 256, 0, stream>>>(sumb, 1024, 0.f);
  DEC_fill<<<4, 256, 0, stream>>>(sqb, 1024, 0.f);
  DEC_stats<<<64, 256, 0, stream>>>(hbuf, EE, 64, sumb, sqb);
  DEC_finalize<<<1, 256, 0, stream>>>(sumb, sqb, c1g2, c1e2, (float)EE, 64, sbuf, tbuf);
  DEC_aggregate<<<(NP*64)/256, 256, 0, stream>>>(hbuf, sbuf, tbuf, 64, cat, 192, 0, x1f);

  // ---- EdgeConv 2 ----
  DEC_norms<<<NP/256, 256, 0, stream>>>(x1f, nrm);
  DEC_knn2<<<BB * (PP/16), 32, 0, stream>>>(cat, 192, nrm, idx2);
  DEC_gather2<<<(EE*64)/256, 256, 0, stream>>>(x1f, idx2, e2);

  DEC_gemm<128><<<dim3(2, EE/32), 256, 0, stream>>>(e2, 128, wc2b, c2b, hbuf, 128, 1);
  DEC_fill<<<4, 256, 0, stream>>>(sumb, 1024, 0.f);
  DEC_fill<<<4, 256, 0, stream>>>(sqb, 1024, 0.f);
  DEC_stats<<<128, 256, 0, stream>>>(hbuf, EE, 128, sumb, sqb);
  DEC_finalize<<<1, 256, 0, stream>>>(sumb, sqb, c2g, c2e, (float)EE, 128, sbuf, tbuf);
  DEC_aggregate<<<(NP*128)/256, 256, 0, stream>>>(hbuf, sbuf, tbuf, 128, cat, 192, 64, (float*)0);

  // ---- l1: 192 -> 1024, BN + global max pool ----
  DEC_gemm<192><<<dim3(16, NP/32), 256, 0, stream>>>(cat, 192, wl1b, l1b, hbuf, 1024, 1);
  DEC_fill<<<4, 256, 0, stream>>>(sumb, 1024, 0.f);
  DEC_fill<<<4, 256, 0, stream>>>(sqb, 1024, 0.f);
  DEC_stats<<<1024, 256, 0, stream>>>(hbuf, NP, 1024, sumb, sqb);
  DEC_bminmax<<<(BB*1024)/256, 256, 0, stream>>>(hbuf, bmx, bmn);
  DEC_pool<<<(BB*1024)/256, 256, 0, stream>>>(sumb, sqb, bmx, bmn, l1g, l1e, pooled);

  // ---- head MLP via WMMA (M padded 16 -> 32 with zero rows) ----
  DEC_pad_bf16<<<(32*1024)/256, 256, 0, stream>>>(pooled, poolb, BB, 1024, 32*1024);
  DEC_gemm<1024><<<dim3(8, 1), 256, 0, stream>>>(poolb, 1024, wm1b, m1b, h1, 512, 1);
  DEC_head_bn<<<2, 256, 0, stream>>>(h1, m1g, m1e, y1, BB, 512);
  DEC_pad_bf16<<<(32*512)/256, 256, 0, stream>>>(y1, y1b, BB, 512, 32*512);
  DEC_gemm<512><<<dim3(4, 1), 256, 0, stream>>>(y1b, 512, wm2b, m2b, h2, 256, 1);
  DEC_head_bn<<<1, 256, 0, stream>>>(h2, m2g, m2e, y2, BB, 256);
  DEC_head_lin<<<1, 256, 0, stream>>>(y2, m3w, m3b, out, BB, 256, 2, 0);
}
